// ClassificationGraphNN_44332652429896
// MI455X (gfx1250) — matrix operand order repaired and checked by reading, not compiled, not run on hardware
//
#include <hip/hip_runtime.h>

typedef __attribute__((ext_vector_type(16))) _Float16 v16h;
typedef __attribute__((ext_vector_type(8)))  _Float16 v8h;
typedef __attribute__((ext_vector_type(8)))  float    v8f;

#define NPTS 1024
#define KNN  20
#define NB   8

// ---------------------------------------------------------------------------
// Weight f32 -> f16 conversion with zero padding to (CoutPad, CinPad)
// ---------------------------------------------------------------------------
__global__ void cvt_weight(const float* __restrict__ Wsrc, const float* __restrict__ bsrc,
                           _Float16* __restrict__ Wdst, float* __restrict__ bdst,
                           int Cout, int Cin, int CoutPad, int CinPad) {
  int idx = blockIdx.x * 256 + threadIdx.x;
  int total = CoutPad * CinPad;
  if (idx < total) {
    int r = idx / CinPad, c = idx % CinPad;
    Wdst[idx] = (r < Cout && c < Cin) ? (_Float16)Wsrc[r * Cin + c] : (_Float16)0.f;
  }
  if (idx < CoutPad) bdst[idx] = (idx < Cout) ? bsrc[idx] : 0.f;
}

// ---------------------------------------------------------------------------
// KNN: per batch, xyz cached in LDS, each thread keeps top-20 by insertion
// ---------------------------------------------------------------------------
__global__ __launch_bounds__(256) void knn_kernel(const float* __restrict__ inp,
                                                  int* __restrict__ nn) {
  __shared__ float xs[NPTS], ys[NPTS], zs[NPTS];
  int b = blockIdx.x, tid = threadIdx.x;
  const float* base = inp + (size_t)b * 3 * NPTS;
  for (int i = tid; i < NPTS; i += 256) {
    xs[i] = base[i]; ys[i] = base[NPTS + i]; zs[i] = base[2 * NPTS + i];
  }
  __syncthreads();
  for (int i = tid; i < NPTS; i += 256) {
    float best[KNN]; int bidx[KNN];
#pragma unroll
    for (int k = 0; k < KNN; ++k) { best[k] = 3.4e38f; bidx[k] = 0; }
    float xi = xs[i], yi = ys[i], zi = zs[i];
    for (int j = 0; j < NPTS; ++j) {
      float dx = xi - xs[j], dy = yi - ys[j], dz = zi - zs[j];
      float d = dx * dx + dy * dy + dz * dz;
      if (d < best[KNN - 1]) {
        int p = KNN - 1;
        while (p > 0 && best[p - 1] > d) { best[p] = best[p - 1]; bidx[p] = bidx[p - 1]; --p; }
        best[p] = d; bidx[p] = j;
      }
    }
    int* o = nn + ((size_t)b * NPTS + i) * KNN;
#pragma unroll
    for (int k = 0; k < KNN; ++k) o[k] = bidx[k];
  }
}

// ---------------------------------------------------------------------------
// WMMA GEMM:  Y[col][row] = relu( W[row][:] dot X[col][:] + bias[row] )
// W: f16 [CoutPad][Cin] row-major (LDS-staged per 32-row tile)
// X: f16 [M][Cin]  (column-contiguous activations)
// Y: f16 [M][CoutPad]
// block = 256 thr = 8 waves (2 row-tiles x 4 col-tiles); grid = (M/64, CoutPad/32)
// ---------------------------------------------------------------------------
__global__ __launch_bounds__(256) void gemm_wmma_relu(
    const _Float16* __restrict__ W, const float* __restrict__ bias,
    const _Float16* __restrict__ X, _Float16* __restrict__ Y,
    int Cin, int CoutPad) {
  extern __shared__ _Float16 ldsW[];
  const int tid = threadIdx.x;
  const int lane = tid & 31;
  const int wid = tid >> 5;
  const int waveRow = wid >> 2;   // 0..1
  const int waveCol = wid & 3;    // 0..3
  const int rowBase = blockIdx.y * 32 + waveRow * 16;
  const long colBase = (long)blockIdx.x * 64 + waveCol * 16;

  // Stage 32 x Cin weight tile into LDS (contiguous chunk), b128 granularity.
  {
    const uint4* src = (const uint4*)(W + (size_t)blockIdx.y * 32 * Cin);
    uint4* dst = (uint4*)ldsW;
    int nvec = (32 * Cin) >> 3;
    for (int i = tid; i < nvec; i += 256) dst[i] = src[i];
  }
  __syncthreads();

  const int laneLo = lane & 15;
  const int laneHi = lane >> 4;
  // B fragment source: this lane's activation column, two 8-half chunks
  const _Float16* xcol = X + (size_t)(colBase + laneLo) * Cin + laneHi * 8;
  // A fragment source: this lane's weight row in LDS, two 8-half chunks
  const _Float16* arow = ldsW + (size_t)(waveRow * 16 + laneLo) * Cin + laneHi * 8;

  v8f acc = {0.f, 0.f, 0.f, 0.f, 0.f, 0.f, 0.f, 0.f};
  for (int k0 = 0; k0 < Cin; k0 += 32) {
    __builtin_prefetch(xcol + k0 + 64, 0, 1);
    v8h a0 = *(const v8h*)(arow + k0);
    v8h a1 = *(const v8h*)(arow + k0 + 16);
    v8h b0 = *(const v8h*)(xcol + k0);
    v8h b1 = *(const v8h*)(xcol + k0 + 16);
    v16h a = __builtin_shufflevector(a0, a1, 0,1,2,3,4,5,6,7,8,9,10,11,12,13,14,15);
    v16h bm = __builtin_shufflevector(b0, b1, 0,1,2,3,4,5,6,7,8,9,10,11,12,13,14,15);
    acc = __builtin_amdgcn_wmma_f32_16x16x32_f16(false, a, false, bm,
                                                 (short)0, acc, false, false);
  }

  const int rowStart = rowBase + laneHi * 8;   // D: lane holds rows rowStart..+7
  v8h out;
#pragma unroll
  for (int j = 0; j < 8; ++j) {
    float v = acc[j] + bias[rowStart + j];
    out[j] = (_Float16)fmaxf(v, 0.f);
  }
  *(v8h*)(Y + (size_t)(colBase + laneLo) * CoutPad + rowStart) = out;
}

// ---------------------------------------------------------------------------
// Block-1 edge input: [e_ij(6) | h_i(3) | h_j(3) | pad] from raw xyz, f32->f16
// ---------------------------------------------------------------------------
__global__ void build_edge1(const float* __restrict__ inp, const int* __restrict__ nn,
                            _Float16* __restrict__ XE) {
  int m = blockIdx.x * 256 + threadIdx.x;        // 0..163839
  int bn = m / KNN;
  int b = bn >> 10, n = bn & 1023;
  int j = nn[m];
  const float* base = inp + (size_t)b * 3 * NPTS;
  _Float16* d = XE + (size_t)m * 32;
#pragma unroll
  for (int c = 0; c < 3; ++c) {
    float hi = base[c * NPTS + n], hj = base[c * NPTS + j];
    d[c]     = (_Float16)hi;
    d[3 + c] = (_Float16)(hi - hj);
    d[6 + c] = (_Float16)hi;
    d[9 + c] = (_Float16)hj;
  }
#pragma unroll
  for (int c = 12; c < 32; ++c) d[c] = (_Float16)0.f;
}

// ---------------------------------------------------------------------------
// Blocks 2/3 edge input: [e_prev(Ce) | h_i(Ch) | h_j(Ch)] (all f16, b128 copies)
// ---------------------------------------------------------------------------
__global__ void build_edge_x(const _Float16* __restrict__ eP, const _Float16* __restrict__ H,
                             const int* __restrict__ nn, _Float16* __restrict__ XE,
                             int Ce, int Ch) {
  int m = blockIdx.x * 256 + threadIdx.x;        // 0..163839
  int bn = m / KNN;
  int b = bn >> 10;
  int jn = (b << 10) + nn[m];
  int Cin = Ce + 2 * Ch;
  const uint4* pe = (const uint4*)(eP + (size_t)m * Ce);
  const uint4* ph = (const uint4*)(H + (size_t)bn * Ch);
  const uint4* pj = (const uint4*)(H + (size_t)jn * Ch);
  uint4* d0 = (uint4*)(XE + (size_t)m * Cin);
  int ve = Ce >> 3, vh = Ch >> 3;
  for (int i = 0; i < ve; ++i) d0[i] = pe[i];
  for (int i = 0; i < vh; ++i) d0[ve + i] = ph[i];
  for (int i = 0; i < vh; ++i) d0[ve + vh + i] = pj[i];
}

// ---------------------------------------------------------------------------
// Node input: [h(Ch) | sum_k e_p(Ce) | pad] ; h source is f32 xyz for block 1
// ---------------------------------------------------------------------------
__global__ void build_node_x(const void* __restrict__ hsrc, int hIsF32,
                             const _Float16* __restrict__ eP, _Float16* __restrict__ XN,
                             int Ch, int Ce, int CinPad, int total) {
  int idx = blockIdx.x * 256 + threadIdx.x;
  if (idx >= total) return;
  int bn = idx / CinPad, c = idx % CinPad;
  float v;
  if (c < Ch) {
    if (hIsF32) {
      int b = bn >> 10, n = bn & 1023;
      v = ((const float*)hsrc)[((size_t)b * 3 + c) * NPTS + n];
    } else {
      v = (float)((const _Float16*)hsrc)[(size_t)bn * Ch + c];
    }
  } else if (c < Ch + Ce) {
    int cc = c - Ch;
    const _Float16* p = eP + (size_t)bn * KNN * Ce + cc;
    float s = 0.f;
#pragma unroll
    for (int k = 0; k < KNN; ++k) s += (float)p[(size_t)k * Ce];
    v = s;
  } else {
    v = 0.f;
  }
  XN[(size_t)bn * CinPad + c] = (_Float16)v;
}

// ---------------------------------------------------------------------------
// feats = concat(h1[64], h2[128], h3[192]) -> [8192][384]
// ---------------------------------------------------------------------------
__global__ void build_feats(const _Float16* __restrict__ H1, const _Float16* __restrict__ H2,
                            const _Float16* __restrict__ H3, _Float16* __restrict__ F) {
  int idx = blockIdx.x * 256 + threadIdx.x;      // 8192*384
  int bn = idx / 384, c = idx % 384;
  _Float16 v;
  if (c < 64)       v = H1[(size_t)bn * 64 + c];
  else if (c < 192) v = H2[(size_t)bn * 128 + (c - 64)];
  else              v = H3[(size_t)bn * 192 + (c - 192)];
  F[idx] = v;
}

// ---------------------------------------------------------------------------
// Global max pool over N per (b, c)
// ---------------------------------------------------------------------------
__global__ void pool_max(const _Float16* __restrict__ FUS, float* __restrict__ pooled) {
  int idx = blockIdx.x * 256 + threadIdx.x;      // 8*512
  if (idx >= NB * 512) return;
  int b = idx / 512, c = idx % 512;
  const _Float16* p = FUS + ((size_t)b * NPTS) * 512 + c;
  float m = -3.4e38f;
  for (int n = 0; n < NPTS; ++n) m = fmaxf(m, (float)p[(size_t)n * 512]);
  pooled[idx] = m;
}

// ---------------------------------------------------------------------------
// Prediction head: 512->256 relu ->128 relu ->40, f32, one block
// ---------------------------------------------------------------------------
__global__ __launch_bounds__(256) void head_kernel(
    const float* __restrict__ pooled,
    const float* __restrict__ w1, const float* __restrict__ b1,
    const float* __restrict__ w2, const float* __restrict__ b2,
    const float* __restrict__ w3, const float* __restrict__ b3,
    float* __restrict__ out) {
  __shared__ float pb[512], t1[256], t2[128];
  int tid = threadIdx.x;
  for (int b = 0; b < NB; ++b) {
    for (int i = tid; i < 512; i += 256) pb[i] = pooled[b * 512 + i];
    __syncthreads();
    {
      float s = b1[tid];
      const float* w = w1 + (size_t)tid * 512;
      for (int i = 0; i < 512; ++i) s += w[i] * pb[i];
      t1[tid] = fmaxf(s, 0.f);
    }
    __syncthreads();
    if (tid < 128) {
      float s = b2[tid];
      const float* w = w2 + (size_t)tid * 256;
      for (int i = 0; i < 256; ++i) s += w[i] * t1[i];
      t2[tid] = fmaxf(s, 0.f);
    }
    __syncthreads();
    if (tid < 40) {
      float s = b3[tid];
      const float* w = w3 + (size_t)tid * 128;
      for (int i = 0; i < 128; ++i) s += w[i] * t2[i];
      out[b * 40 + tid] = s;
    }
    __syncthreads();
  }
}

// ---------------------------------------------------------------------------
struct MSpec { int wi, bi, Cout, Cin, CoutPad, CinPad; };

extern "C" void kernel_launch(void* const* d_in, const int* in_sizes, int n_in,
                              void* d_out, int out_size, void* d_ws, size_t ws_size,
                              hipStream_t stream) {
  (void)in_sizes; (void)n_in; (void)out_size; (void)ws_size;
  static const MSpec specs[13] = {
      {1, 2, 32, 12, 32, 32},     // he1
      {3, 4, 64, 32, 64, 32},     // he2
      {5, 6, 32, 67, 32, 96},     // hn1
      {7, 8, 64, 32, 64, 32},     // hn2
      {9, 10, 48, 192, 64, 192},  // b1e1
      {11, 12, 128, 48, 128, 64}, // b1e2
      {13, 14, 48, 192, 64, 192}, // b1n1
      {15, 16, 128, 48, 128, 64}, // b1n2
      {17, 18, 96, 384, 96, 384}, // b2e1
      {19, 20, 192, 96, 192, 96}, // b2e2
      {21, 22, 96, 320, 96, 320}, // b2n1
      {23, 24, 192, 96, 192, 96}, // b2n2
      {25, 26, 512, 384, 512, 384}// f
  };

  char* ws = (char*)d_ws;
  size_t off = 0;
  auto alloc = [&](size_t bytes) -> char* {
    char* p = ws + off;
    off = (off + bytes + 255) & ~(size_t)255;
    return p;
  };

  const int ME = NB * NPTS * KNN;  // 163840
  const int MN = NB * NPTS;        // 8192

  int* nnIdx = (int*)alloc((size_t)ME * 4);
  _Float16* W16[13]; float* B32[13];
  for (int i = 0; i < 13; ++i) {
    W16[i] = (_Float16*)alloc((size_t)specs[i].CoutPad * specs[i].CinPad * 2);
    B32[i] = (float*)alloc((size_t)specs[i].CoutPad * 4);
  }
  _Float16* XE   = (_Float16*)alloc((size_t)ME * 384 * 2);
  _Float16* MIDE = (_Float16*)alloc((size_t)ME * 96 * 2);
  _Float16* EA   = (_Float16*)alloc((size_t)ME * 192 * 2);
  _Float16* EB   = (_Float16*)alloc((size_t)ME * 192 * 2);
  _Float16* XN   = (_Float16*)alloc((size_t)MN * 384 * 2);
  _Float16* MIDN = (_Float16*)alloc((size_t)MN * 96 * 2);
  _Float16* H1   = (_Float16*)alloc((size_t)MN * 64 * 2);
  _Float16* H2   = (_Float16*)alloc((size_t)MN * 128 * 2);
  _Float16* H3   = (_Float16*)alloc((size_t)MN * 192 * 2);
  _Float16* FE   = (_Float16*)alloc((size_t)MN * 384 * 2);
  _Float16* FUS  = (_Float16*)alloc((size_t)MN * 512 * 2);
  float* POOL    = (float*)alloc((size_t)NB * 512 * 4);

  const float* inp = (const float*)d_in[0];

  // 1) convert + pad weights to f16
  for (int i = 0; i < 13; ++i) {
    int total = specs[i].CoutPad * specs[i].CinPad;
    cvt_weight<<<(total + 255) / 256, 256, 0, stream>>>(
        (const float*)d_in[specs[i].wi], (const float*)d_in[specs[i].bi],
        W16[i], B32[i], specs[i].Cout, specs[i].Cin, specs[i].CoutPad, specs[i].CinPad);
  }

  // 2) KNN graph
  knn_kernel<<<NB, 256, 0, stream>>>(inp, nnIdx);

  auto gemm = [&](int i, const _Float16* X, _Float16* Y, int M) {
    dim3 g(M / 64, specs[i].CoutPad / 32);
    size_t lds = (size_t)32 * specs[i].CinPad * 2;
    gemm_wmma_relu<<<g, 256, lds, stream>>>(W16[i], B32[i], X, Y,
                                            specs[i].CinPad, specs[i].CoutPad);
  };

  // 3) MP block 1
  build_edge1<<<ME / 256, 256, 0, stream>>>(inp, nnIdx, XE);
  gemm(0, XE, MIDE, ME);                       // 12(32) -> 32
  gemm(1, MIDE, EA, ME);                       // 32 -> 64  (e1)
  {
    int tot = MN * 96;
    build_node_x<<<(tot + 255) / 256, 256, 0, stream>>>(inp, 1, EA, XN, 3, 64, 96, tot);
  }
  gemm(2, XN, MIDN, MN);                       // 67(96) -> 32
  gemm(3, MIDN, H1, MN);                       // 32 -> 64  (h1)

  // 4) MP block 2
  build_edge_x<<<ME / 256, 256, 0, stream>>>(EA, H1, nnIdx, XE, 64, 64);
  gemm(4, XE, MIDE, ME);                       // 192 -> 48(64)
  gemm(5, MIDE, EB, ME);                       // 48(64) -> 128 (e2)
  {
    int tot = MN * 192;
    build_node_x<<<(tot + 255) / 256, 256, 0, stream>>>(H1, 0, EB, XN, 64, 128, 192, tot);
  }
  gemm(6, XN, MIDN, MN);                       // 192 -> 48(64)
  gemm(7, MIDN, H2, MN);                       // 48(64) -> 128 (h2)

  // 5) MP block 3
  build_edge_x<<<ME / 256, 256, 0, stream>>>(EB, H2, nnIdx, XE, 128, 128);
  gemm(8, XE, MIDE, ME);                       // 384 -> 96
  gemm(9, MIDE, EA, ME);                       // 96 -> 192 (e3)
  {
    int tot = MN * 320;
    build_node_x<<<(tot + 255) / 256, 256, 0, stream>>>(H2, 0, EA, XN, 128, 192, 320, tot);
  }
  gemm(10, XN, MIDN, MN);                      // 320 -> 96
  gemm(11, MIDN, H3, MN);                      // 96 -> 192 (h3)

  // 6) fuse + pool + head
  {
    int tot = MN * 384;
    build_feats<<<(tot + 255) / 256, 256, 0, stream>>>(H1, H2, H3, FE);
  }
  gemm(12, FE, FUS, MN);                       // 384 -> 512
  pool_max<<<(NB * 512 + 255) / 256, 256, 0, stream>>>(FUS, POOL);
  head_kernel<<<1, 256, 0, stream>>>(POOL,
      (const float*)d_in[27], (const float*)d_in[28],
      (const float*)d_in[29], (const float*)d_in[30],
      (const float*)d_in[31], (const float*)d_in[32],
      (float*)d_out);

  // 7) second output: inputs passthrough
  hipMemcpyAsync((float*)d_out + NB * 40, d_in[0],
                 (size_t)NB * 3 * NPTS * sizeof(float),
                 hipMemcpyDeviceToDevice, stream);
}
